// CrossAttention_14139032338690
// MI455X (gfx1250) — compile-verified, matrix operand order checked by reading, not run on hardware
//
#include <hip/hip_runtime.h>
#include <hip/hip_bf16.h>
#include <cstdint>

// ---------------------------------------------------------------------------
// CrossAttention for MI455X (gfx1250, wave32, WMMA + TDM double-buffering).
// Pipeline:
//   0) cvt_f32_bf16 x4     : Wq/Wk/Wv/Wo -> bf16 in ws (one-shot)
//   1) gemm_bf16<float,1>  : Q = x @ Wq^T  (+rmsnorm+rope(32)+1/sqrt(D)) -> bf16 [B,H,T,D]
//   2) gemm_bf16<float,2>  : K = ctx @ Wk^T (+rmsnorm+rope(1000))        -> bf16 [B,H,S,D]
//   3) gemm_bf16<float,3>  : V = ctx @ Wv^T                              -> bf16 [B,H,S,D]
//   4) attn_kernel         : softmax(QK^T)V                              -> bf16 [B,T,H,D]
//   5) gemm_bf16<bf16,0>   : out = O @ Wo^T                              -> f32
// W tiles (and A tiles when A is bf16) are staged into LDS by the Tensor
// Data Mover with double-buffered tiles overlapped with WMMA compute.
// Workspace: Q 16MB | K 96MB | V 96MB | O 16MB | Wbf 4x2MB  (232 MB)
// ---------------------------------------------------------------------------

#define DIMC 1024
#define NH   16
#define HD   64
#define TT   64
#define SSQ  384

typedef __attribute__((ext_vector_type(16))) __bf16 v16bf;
typedef __attribute__((ext_vector_type(8)))  float  v8f;
typedef unsigned int __attribute__((ext_vector_type(4))) v4u;
typedef int          __attribute__((ext_vector_type(4))) v4i;
typedef int          __attribute__((ext_vector_type(8))) v8i;

__device__ __forceinline__ __bf16 f2bf(float f) {
  union { float f; uint32_t u; } in; in.f = f;
  uint32_t u = in.u;
  uint32_t r = (u + 0x7FFFu + ((u >> 16) & 1u)) >> 16;   // round-to-nearest-even
  union { uint16_t s; __bf16 b; } out; out.s = (uint16_t)r;
  return out.b;
}
__device__ __forceinline__ float to_f(float x) { return x; }
__device__ __forceinline__ float to_f(__bf16 x) {
  union { uint16_t s; __bf16 b; } c; c.b = x;
  union { uint32_t u; float f; } o; o.u = ((uint32_t)c.s) << 16;
  return o.f;
}

// Pack two f32 -> bf16x2 dword (v_cvt_pk_bf16_f32 when available).
__device__ __forceinline__ uint32_t pack2bf(float lo, float hi) {
#if __has_builtin(__builtin_amdgcn_cvt_pk_bf16_f32)
  typedef __attribute__((ext_vector_type(2))) __bf16 v2bf;
  union { v2bf v; uint32_t u; } cv;
  cv.v = __builtin_amdgcn_cvt_pk_bf16_f32(lo, hi);
  return cv.u;
#else
  union { __bf16 b[2]; uint32_t u; } cv;
  cv.b[0] = f2bf(lo); cv.b[1] = f2bf(hi);
  return cv.u;
#endif
}

// One 16-lane slice of a 16x32 bf16 WMMA fragment: 8 contiguous halves at
// p[0..7] and 8 at p[16..23] (CDNA5 16-bit A/B VGPR striping).
__device__ __forceinline__ v16bf frag16(const __bf16* p) {
  union { v16bf v; uint4 u[2]; } f;
  f.u[0] = *(const uint4*)(p);
  f.u[1] = *(const uint4*)(p + 16);
  return f.v;
}

// ---------------------------------------------------------------------------
// Tensor Data Mover: DMA a 2D bf16 tile (tileD1 rows x tileD0 cols) from a
// row-major tensor (row stride strideD0 elems) into contiguous LDS.
// D# per CDNA5 ISA (groups 2/3 zeroed => 2D tensor).
// lds_addr = low 32 bits of generic pointer (flat-LDS aperture truncation).
// ---------------------------------------------------------------------------
#if __has_builtin(__builtin_amdgcn_tensor_load_to_lds)
#define HAVE_TDM 1
#else
#define HAVE_TDM 0
#endif

__device__ __forceinline__ void tdm_load_2d_bf16(
    void* ldsDst, const void* gsrc,
    uint32_t tensorD0, uint32_t tensorD1,
    uint32_t tileD0, uint32_t tileD1, uint32_t strideD0)
{
#if HAVE_TDM
  uint64_t ga = (uint64_t)(uintptr_t)gsrc;
  uint32_t la = (uint32_t)(uintptr_t)ldsDst;

  v4u g0 = {};
  g0[0] = 1u;                                                // count=1 (user D#)
  g0[1] = la;                                                // lds_addr
  g0[2] = (uint32_t)ga;                                      // global_addr[31:0]
  g0[3] = (uint32_t)((ga >> 32) & 0x01FFFFFFu) | (2u << 30); // ga[56:32], type=2

  v8i g1 = {};
  g1[0] = (int)(1u << 16);                                   // data_size=1 (2B)
  g1[1] = (int)((tensorD0 & 0xFFFFu) << 16);                 // tensor_dim0 lo
  g1[2] = (int)(((tensorD0 >> 16) & 0xFFFFu) |
                ((tensorD1 & 0xFFFFu) << 16));               // td0 hi | td1 lo
  g1[3] = (int)(((tensorD1 >> 16) & 0xFFFFu) |
                ((tileD0 & 0xFFFFu) << 16));                 // td1 hi | tile_dim0
  g1[4] = (int)(tileD1 & 0xFFFFu);                           // tile_dim1 (dim2=0)
  g1[5] = (int)strideD0;                                     // dim0_stride lo32
  g1[6] = 0;
  g1[7] = 0;

  v4i gz = {};
#if defined(__clang_major__) && (__clang_major__ >= 23)
  v8i gz8 = {};
  __builtin_amdgcn_tensor_load_to_lds(g0, g1, gz, gz, gz8, 0);
#else
  __builtin_amdgcn_tensor_load_to_lds(g0, g1, gz, gz, 0);
#endif
#endif
}

__device__ __forceinline__ void wait_tensor0() {
#if __has_builtin(__builtin_amdgcn_s_wait_tensorcnt)
  __builtin_amdgcn_s_wait_tensorcnt(0);
#endif
}
__device__ __forceinline__ void wait_tensor1() {
#if __has_builtin(__builtin_amdgcn_s_wait_tensorcnt)
  __builtin_amdgcn_s_wait_tensorcnt(1);
#endif
}
__device__ __forceinline__ void wait_tensor2() {
#if __has_builtin(__builtin_amdgcn_s_wait_tensorcnt)
  __builtin_amdgcn_s_wait_tensorcnt(2);
#endif
}

// ---------------------------------------------------------------------------
// One-shot f32 -> bf16 conversion (weights). 8 elems/thread, b128 in/out.
// ---------------------------------------------------------------------------
__global__ __launch_bounds__(256) void cvt_f32_bf16(
    const float* __restrict__ src, __bf16* __restrict__ dst, int n)
{
  int i = (blockIdx.x * 256 + threadIdx.x) * 8;
  if (i >= n) return;
  float4 a = *(const float4*)(src + i);
  float4 b = *(const float4*)(src + i + 4);
  uint4 o;
  o.x = pack2bf(a.x, a.y); o.y = pack2bf(a.z, a.w);
  o.z = pack2bf(b.x, b.y); o.w = pack2bf(b.z, b.w);
  *(uint4*)(dst + i) = o;
}

// ---------------------------------------------------------------------------
// bf16 GEMM: C[M,1024] = A[M,1024] @ Wb[1024,1024]^T   (Wb already bf16).
// Block tile 128x64, 256 threads = 8 waves; wave w owns rows [16w,16w+16) x
// all 64 cols: one A-frag reused across 4 B-frags -> 4 wmma per K-step.
// W tiles (and A tiles when AT==bf16) staged by TDM, double-buffered.
// MODE 0: f32 out   MODE 1: Q epilogue   MODE 2: K epilogue   MODE 3: V.
// ---------------------------------------------------------------------------
template<typename AT, int MODE>
__global__ __launch_bounds__(256) void gemm_bf16(
    const AT* __restrict__ A, const __bf16* __restrict__ Wb,
    float* __restrict__ Cout, __bf16* __restrict__ Obf,
    const float* __restrict__ normw, int M)
{
  __shared__ __align__(16) __bf16 As[2][128][32];   // 16 KB (double buffer)
  __shared__ __align__(16) __bf16 Bs[2][64][32];    //  8 KB (double buffer)
  __shared__ __align__(16) float  Cs[128][64];      // 32 KB

  const int tid  = threadIdx.x;
  const int lane = tid & 31;
  const int wave = tid >> 5;
  const int m0   = blockIdx.x * 128;
  const int n0   = blockIdx.y * 64;

  const int fr  = lane & 15;
  const int kof = (lane & 16) ? 8 : 0;
  const int mo  = (lane < 16) ? 0 : 8;

  constexpr bool A_BF16 = (sizeof(AT) == 2);
  constexpr bool TDM    = (HAVE_TDM != 0);

  // manual staging assignments
  const int arow = tid >> 1;            // 128 rows, 2 threads/row, 16 elems
  const int akof = (tid & 1) * 16;
  const int brow = tid >> 2;            // 64 rows, 4 threads/row, 8 elems
  const int bkof = (tid & 3) * 8;

  v8f acc[4] = {{}, {}, {}, {}};

  // TDM prologue: bring tile kb=0 into buffer 0 (1 or 2 descriptors/tile).
  if (TDM && wave == 0) {
    tdm_load_2d_bf16(&Bs[0][0][0], Wb + (size_t)n0 * DIMC,
                     DIMC, DIMC, 32, 64, DIMC);
    if (A_BF16)
      tdm_load_2d_bf16(&As[0][0][0], (const void*)(A + (size_t)m0 * DIMC),
                       DIMC, (uint32_t)M, 32, 128, DIMC);
  }

  for (int kb = 0; kb < DIMC; kb += 32) {
    const int buf = (kb >> 5) & 1;

    // Issue next tile's DMA, then wait for the current tile only
    // (TENSORcnt completes in-order per wave -> wait <= #newly-issued).
    if (TDM && wave == 0) {
      if (kb + 32 < DIMC) {
        tdm_load_2d_bf16(&Bs[buf ^ 1][0][0],
                         Wb + (size_t)n0 * DIMC + kb + 32,
                         DIMC, DIMC, 32, 64, DIMC);
        if (A_BF16) {
          tdm_load_2d_bf16(&As[buf ^ 1][0][0],
                           (const void*)(A + (size_t)m0 * DIMC + kb + 32),
                           DIMC, (uint32_t)M, 32, 128, DIMC);
          wait_tensor2();
        } else {
          wait_tensor1();
        }
      } else {
        wait_tensor0();
      }
    }

    // Manual A staging (f32 source, or no-TDM fallback).
    if (!(TDM && A_BF16)) {
      const AT* ap = A + (size_t)(m0 + arow) * DIMC + kb + akof;
      if (kb + 32 < DIMC) __builtin_prefetch(ap + 32, 0, 1);  // global_prefetch_b8
      if (A_BF16) {
        *(uint4*)&As[buf][arow][akof]     = *(const uint4*)(ap);
        *(uint4*)&As[buf][arow][akof + 8] = *(const uint4*)(ap + 8);
      } else {
        #pragma unroll
        for (int j = 0; j < 16; j += 2)
          *(uint32_t*)&As[buf][arow][akof + j] =
              pack2bf(to_f(ap[j]), to_f(ap[j + 1]));
      }
    }
    // Manual B staging only without TDM.
    if (!TDM) {
      const __bf16* wp = Wb + (size_t)(n0 + brow) * DIMC + kb + bkof;
      *(uint4*)&Bs[buf][brow][bkof] = *(const uint4*)(wp);
    }

    __syncthreads();

    v16bf a = frag16(&As[buf][wave * 16 + fr][kof]);
    #pragma unroll
    for (int nt = 0; nt < 4; ++nt) {
      v16bf b = frag16(&Bs[buf][nt * 16 + fr][kof]);
      acc[nt] = __builtin_amdgcn_wmma_f32_16x16x32_bf16(
          false, a, false, b, (short)0, acc[nt], false, false);
    }
    __syncthreads();
  }

  // Scatter accumulators to LDS so epilogues see full 64-wide head rows.
  #pragma unroll
  for (int nt = 0; nt < 4; ++nt)
    #pragma unroll
    for (int r = 0; r < 8; ++r)
      Cs[wave * 16 + mo + r][nt * 16 + fr] = acc[nt][r];
  __syncthreads();

  if (MODE == 0) {
    for (int i = tid; i < 128 * 64; i += 256) {
      int r = i >> 6, cc = i & 63;
      Cout[(size_t)(m0 + r) * DIMC + n0 + cc] = Cs[r][cc];
    }
  } else if (MODE == 3) {
    for (int i = tid; i < 128 * 64; i += 256) {
      int r = i >> 6, d = i & 63;
      int m = m0 + r;
      int b = m / SSQ, s = m % SSQ;
      Obf[(((size_t)b * NH + blockIdx.y) * SSQ + s) * HD + d] = f2bf(Cs[r][d]);
    }
  } else if (tid < 128) {               // MODE 1 / 2: rmsnorm + rope per row
    const int   r      = tid;
    const int   m      = m0 + r;
    const int   h      = blockIdx.y;
    const int   seqlen = (MODE == 1) ? TT : SSQ;
    const int   b      = m / seqlen;
    const int   pos    = m % seqlen;
    const float theta  = (MODE == 1) ? 32.0f : 1000.0f;
    const float osc    = (MODE == 1) ? 0.125f : 1.0f;   // fold D^-0.5 into Q

    float v[64];
    float ssq = 0.f;
    #pragma unroll
    for (int d = 0; d < 64; ++d) { v[d] = Cs[r][d]; ssq += v[d] * v[d]; }
    float rms = rsqrtf(ssq * (1.0f / 64.0f) + 1e-6f);
    const float lt = -logf(theta) * (1.0f / 32.0f);

    __bf16* orow = Obf + (((size_t)b * NH + h) * seqlen + pos) * HD;
    #pragma unroll
    for (int j = 0; j < 32; ++j) {
      float ang = (float)pos * __expf(lt * (float)j);
      float sn, cs; __sincosf(ang, &sn, &cs);
      float x1 = v[j]      * rms * normw[j];
      float x2 = v[j + 32] * rms * normw[j + 32];
      orow[j]      = f2bf((x1 * cs - x2 * sn) * osc);
      orow[j + 32] = f2bf((x2 * cs + x1 * sn) * osc);
    }
  }
}

// ---------------------------------------------------------------------------
// Attention: one block per (b,h). 256 threads = 8 waves.
// Dynamic LDS: 96KB f32 scores (reused in place for bf16 P) + 48KB V^T.
// ---------------------------------------------------------------------------
__global__ __launch_bounds__(256) void attn_kernel(
    const __bf16* __restrict__ Qb, const __bf16* __restrict__ Kb,
    const __bf16* __restrict__ Vb, __bf16* __restrict__ Ob)
{
  extern __shared__ __align__(16) char smem[];
  float*  Ss = (float*)smem;                       // [64][384] f32
  __bf16* Pb = (__bf16*)smem;                      // row r at r*768 (in-place)
  __bf16* VT = (__bf16*)(smem + 64 * SSQ * 4);     // [64][384] bf16 = V^T

  const int tid  = threadIdx.x;
  const int lane = tid & 31;
  const int wave = tid >> 5;
  const int bh = blockIdx.x;
  const int b  = bh >> 4, h = bh & 15;

  const __bf16* qp = Qb + (size_t)bh * TT  * HD;
  const __bf16* kp = Kb + (size_t)bh * SSQ * HD;
  const __bf16* vp = Vb + (size_t)bh * SSQ * HD;

  const int fr  = lane & 15;
  const int kof = (lane & 16) ? 8 : 0;
  const int mo  = (lane < 16) ? 0 : 8;

  // ---- Phase 1: S = Q K^T (scale folded into Q) -> LDS f32 ----
  {
    const int rg   = wave & 3;        // 16-row group
    const int half = wave >> 2;       // 192-col half
    const int m    = rg * 16 + fr;
    v16bf a0 = frag16(qp + m * HD + kof);
    v16bf a1 = frag16(qp + m * HD + 32 + kof);
    for (int nt = 0; nt < 12; ++nt) {
      int s0 = half * 192 + nt * 16;
      v16bf b0 = frag16(kp + (s0 + fr) * HD + kof);
      v16bf b1 = frag16(kp + (s0 + fr) * HD + 32 + kof);
      v8f c = {};
      c = __builtin_amdgcn_wmma_f32_16x16x32_bf16(false, a0, false, b0, (short)0, c, false, false);
      c = __builtin_amdgcn_wmma_f32_16x16x32_bf16(false, a1, false, b1, (short)0, c, false, false);
      #pragma unroll
      for (int r = 0; r < 8; ++r)
        Ss[(rg * 16 + mo + r) * SSQ + s0 + fr] = c[r];
    }
  }
  __syncthreads();

  // ---- Phase 2: softmax (waves 0-1, one thread/row, in-place bf16 P);
  //               waves 2-7 stage V^T into LDS concurrently ----
  if (tid < 64) {
    float* row = Ss + tid * SSQ;
    float mx = row[0];
    for (int i = 1; i < SSQ; ++i) mx = fmaxf(mx, row[i]);
    float sum = 0.f;
    for (int i = 0; i < SSQ; ++i) { float e = __expf(row[i] - mx); row[i] = e; sum += e; }
    float inv = 1.0f / sum;
    __bf16* prow = Pb + tid * 768;    // ascending write offset <= read offset
    for (int i = 0; i < SSQ; ++i) prow[i] = f2bf(row[i] * inv);
  } else {
    for (int idx = tid - 64; idx < SSQ * HD; idx += 192) {
      int s = idx >> 6, d = idx & 63;
      VT[d * SSQ + s] = vp[idx];
    }
  }
  __syncthreads();

  // ---- Phase 3: O = P V -> bf16 [B,T,H,D] ----
  {
    const int rg = wave & 3;
    const int dh = wave >> 2;         // 32-wide D half
    const int m  = rg * 16 + fr;
    v8f c0 = {}; v8f c1 = {};
    for (int sc = 0; sc < 12; ++sc) {
      int k0 = sc * 32 + kof;
      v16bf a  = frag16(Pb + m * 768 + k0);
      v16bf b0 = frag16(VT + (dh * 32 + fr) * SSQ + k0);
      v16bf b1 = frag16(VT + (dh * 32 + 16 + fr) * SSQ + k0);
      c0 = __builtin_amdgcn_wmma_f32_16x16x32_bf16(false, a, false, b0, (short)0, c0, false, false);
      c1 = __builtin_amdgcn_wmma_f32_16x16x32_bf16(false, a, false, b1, (short)0, c1, false, false);
    }
    #pragma unroll
    for (int r = 0; r < 8; ++r) {
      int t = rg * 16 + mo + r;
      size_t base = (((size_t)b * TT + t) * NH + h) * HD;
      Ob[base + dh * 32 + fr]      = f2bf(c0[r]);
      Ob[base + dh * 32 + 16 + fr] = f2bf(c1[r]);
    }
  }
}

// ---------------------------------------------------------------------------
extern "C" void kernel_launch(void* const* d_in, const int* in_sizes, int n_in,
                              void* d_out, int out_size, void* d_ws, size_t ws_size,
                              hipStream_t stream) {
  const float* x   = (const float*)d_in[0];
  const float* ctx = (const float*)d_in[1];
  const float* Wq  = (const float*)d_in[2];
  const float* Wk  = (const float*)d_in[3];
  const float* Wv  = (const float*)d_in[4];
  const float* Wo  = (const float*)d_in[5];
  const float* qnw = (const float*)d_in[6];
  const float* knw = (const float*)d_in[7];
  float* out = (float*)d_out;

  char* ws = (char*)d_ws;
  const size_t MB = 1024 * 1024;
  __bf16* Qb  = (__bf16*)(ws);               // 16 MB
  __bf16* Kb  = (__bf16*)(ws + 16  * MB);    // 96 MB
  __bf16* Vb  = (__bf16*)(ws + 112 * MB);    // 96 MB
  __bf16* Ob  = (__bf16*)(ws + 208 * MB);    // 16 MB
  __bf16* Wqb = (__bf16*)(ws + 224 * MB);    //  2 MB
  __bf16* Wkb = (__bf16*)(ws + 226 * MB);    //  2 MB
  __bf16* Wvb = (__bf16*)(ws + 228 * MB);    //  2 MB
  __bf16* Wob = (__bf16*)(ws + 230 * MB);    //  2 MB

  const int nW = DIMC * DIMC;                // 1M elems, 512 blocks
  cvt_f32_bf16<<<512, 256, 0, stream>>>(Wq, Wqb, nW);
  cvt_f32_bf16<<<512, 256, 0, stream>>>(Wk, Wkb, nW);
  cvt_f32_bf16<<<512, 256, 0, stream>>>(Wv, Wvb, nW);
  cvt_f32_bf16<<<512, 256, 0, stream>>>(Wo, Wob, nW);

  gemm_bf16<float, 1><<<dim3(64,  16), 256, 0, stream>>>(x,   Wqb, nullptr, Qb, qnw, 8192);
  gemm_bf16<float, 2><<<dim3(384, 16), 256, 0, stream>>>(ctx, Wkb, nullptr, Kb, knw, 49152);
  gemm_bf16<float, 3><<<dim3(384, 16), 256, 0, stream>>>(ctx, Wvb, nullptr, Vb, nullptr, 49152);

  const size_t smem = (size_t)64 * SSQ * 4 + (size_t)64 * SSQ * 2;   // 147456 B
  hipFuncSetAttribute((const void*)attn_kernel,
                      hipFuncAttributeMaxDynamicSharedMemorySize, (int)smem);
  attn_kernel<<<2048, 256, smem, stream>>>(Qb, Kb, Vb, Ob);

  gemm_bf16<__bf16, 0><<<dim3(64, 16), 256, 0, stream>>>(Ob, Wob, out, nullptr, nullptr, 8192);
}
